// SelfMHA_55765855371430
// MI455X (gfx1250) — compile-verified, matrix operand order checked by reading, not run on hardware
//
#include <hip/hip_runtime.h>
#include <hip/hip_bf16.h>

typedef __bf16 bf16_t;
typedef __attribute__((ext_vector_type(16))) __bf16 v16bf;
typedef __attribute__((ext_vector_type(8)))  __bf16 v8bf;
typedef __attribute__((ext_vector_type(8)))  float  v8f;

#define S_LEN 2048
#define HIDW  2048
#define NH    16
#define NKV   8
#define HD    128

// Build a 16x16x32 bf16 A/B fragment from two contiguous 8-element (16B) chunks.
__device__ __forceinline__ v16bf load_frag(const bf16_t* lo, const bf16_t* hi) {
  v8bf a = *(const v8bf*)lo;
  v8bf b = *(const v8bf*)hi;
  v16bf r;
#pragma unroll
  for (int i = 0; i < 8; ++i) { r[i] = a[i]; r[8 + i] = b[i]; }
  return r;
}

// ---------------- f32 -> bf16 convert ----------------
__global__ void k_f2bf(const float* __restrict__ src, bf16_t* __restrict__ dst, int n) {
  int i = blockIdx.x * blockDim.x + threadIdx.x;
  if (i < n) dst[i] = (bf16_t)src[i];
}

// ---------------- GEMM: C[M,N] = A[M,K] * B[N,K]^T (both K-contiguous bf16, f32 out)
// One wave computes a 32x64 tile (2 A row-blocks x 4 B col-blocks):
// 8 WMMAs per 12 b128 loads per K=32 step. grid = (N/64, M/32), block = 32.
__global__ void k_gemm_nt(const bf16_t* __restrict__ A, const bf16_t* __restrict__ B,
                          float* __restrict__ C, int M, int N, int K) {
  const int lane = threadIdx.x;
  const int row  = lane & 15;
  const int half = lane >> 4;
  const int m0 = blockIdx.y * 32;
  const int n0 = blockIdx.x * 64;
  v8f acc[2][4] = {};
  const bf16_t* ap0 = A + (size_t)(m0 + row) * K + half * 8;
  const bf16_t* ap1 = ap0 + (size_t)16 * K;
  const bf16_t* bp  = B + (size_t)(n0 + row) * K + half * 8;
  const size_t bstep = (size_t)16 * K;
  for (int k = 0; k < K; k += 32) {
    v16bf a0 = load_frag(ap0 + k, ap0 + k + 16);
    v16bf a1 = load_frag(ap1 + k, ap1 + k + 16);
    v16bf b0 = load_frag(bp + k,             bp + k + 16);
    v16bf b1 = load_frag(bp + bstep + k,     bp + bstep + k + 16);
    v16bf b2 = load_frag(bp + 2 * bstep + k, bp + 2 * bstep + k + 16);
    v16bf b3 = load_frag(bp + 3 * bstep + k, bp + 3 * bstep + k + 16);
    acc[0][0] = __builtin_amdgcn_wmma_f32_16x16x32_bf16(false, a0, false, b0, (short)0, acc[0][0], false, false);
    acc[0][1] = __builtin_amdgcn_wmma_f32_16x16x32_bf16(false, a0, false, b1, (short)0, acc[0][1], false, false);
    acc[0][2] = __builtin_amdgcn_wmma_f32_16x16x32_bf16(false, a0, false, b2, (short)0, acc[0][2], false, false);
    acc[0][3] = __builtin_amdgcn_wmma_f32_16x16x32_bf16(false, a0, false, b3, (short)0, acc[0][3], false, false);
    acc[1][0] = __builtin_amdgcn_wmma_f32_16x16x32_bf16(false, a1, false, b0, (short)0, acc[1][0], false, false);
    acc[1][1] = __builtin_amdgcn_wmma_f32_16x16x32_bf16(false, a1, false, b1, (short)0, acc[1][1], false, false);
    acc[1][2] = __builtin_amdgcn_wmma_f32_16x16x32_bf16(false, a1, false, b2, (short)0, acc[1][2], false, false);
    acc[1][3] = __builtin_amdgcn_wmma_f32_16x16x32_bf16(false, a1, false, b3, (short)0, acc[1][3], false, false);
  }
#pragma unroll
  for (int blk = 0; blk < 2; ++blk) {
#pragma unroll
    for (int r = 0; r < 8; ++r) {
      const int m = m0 + blk * 16 + r + 8 * half;
      float* c = C + (size_t)m * N + n0 + row;
      c[0]  = acc[blk][0][r];
      c[16] = acc[blk][1][r];
      c[32] = acc[blk][2][r];
      c[48] = acc[blk][3][r];
    }
  }
}

// ---------------- RoPE + pack: qkv f32 [S,4096] -> q bf16 [H,S,D], k bf16 [KV,S,D], vT bf16 [KV,D,S]
__global__ void k_rope_pack(const float* __restrict__ qkv,
                            const float* __restrict__ cosb, const float* __restrict__ sinb,
                            bf16_t* __restrict__ qb, bf16_t* __restrict__ kb, bf16_t* __restrict__ vT) {
  const int idx = blockIdx.x * blockDim.x + threadIdx.x;
  if (idx >= S_LEN * 4096) return;
  const int s = idx >> 12;      // / 4096
  const int c = idx & 4095;
  if (c < HIDW) {               // Q
    const int h = c >> 7, d = c & 127;
    const float x = qkv[idx];
    const float p = qkv[(size_t)s * 4096 + (c ^ 64)];   // rotate-half partner
    const float cs = cosb[s * HD + d], sn = sinb[s * HD + d];
    const float r = (d < 64) ? (x * cs - p * sn) : (x * cs + p * sn);
    qb[((size_t)h * S_LEN + s) * HD + d] = (bf16_t)r;
  } else if (c < HIDW + NKV * HD) {  // K
    const int cc = c - HIDW;
    const int h = cc >> 7, d = cc & 127;
    const float x = qkv[idx];
    const float p = qkv[(size_t)s * 4096 + (c ^ 64)];
    const float cs = cosb[s * HD + d], sn = sinb[s * HD + d];
    const float r = (d < 64) ? (x * cs - p * sn) : (x * cs + p * sn);
    kb[((size_t)h * S_LEN + s) * HD + d] = (bf16_t)r;
  } else {                      // V, transposed to [kv][d][s]
    const int cc = c - HIDW - NKV * HD;
    const int h = cc >> 7, d = cc & 127;
    vT[((size_t)h * HD + d) * S_LEN + s] = (bf16_t)qkv[idx];
  }
}

// ---------------- causal GQA flash attention, one wave per (head, 16-query block)
__global__ void k_attn(const bf16_t* __restrict__ qb, const bf16_t* __restrict__ kb,
                       const bf16_t* __restrict__ vT, bf16_t* __restrict__ ctxb) {
  __shared__ __align__(16) bf16_t Plds[16 * 32];
  const int lane = threadIdx.x;
  const int col  = lane & 15;    // A: M row  |  B: N col  |  C: N col
  const int half = lane >> 4;
  const int qi = blockIdx.x & ((S_LEN / 16) - 1);
  const int h  = blockIdx.x >> 7;           // S/16 == 128
  const int kv = h >> 1;                    // H/HKV == 2
  const float sc = 0.08838834764831845f;    // 1/sqrt(128)

  // Q fragments for K-dim 0..127 (4 chunks of 32)
  const bf16_t* Qp = qb + ((size_t)h * S_LEN + qi * 16 + col) * HD + half * 8;
  v16bf aq[4];
#pragma unroll
  for (int c4 = 0; c4 < 4; ++c4)
    aq[c4] = load_frag(Qp + c4 * 32, Qp + c4 * 32 + 16);

  float mrow[8], lrow[8];
#pragma unroll
  for (int r = 0; r < 8; ++r) { mrow[r] = -3.0e38f; lrow[r] = 0.0f; }
  v8f acc[8] = {};

  const int nch = (qi >> 1) + 1;            // 32-key chunks needed for causal coverage
  const bf16_t* Kbase = kb + (size_t)kv * S_LEN * HD;
  const bf16_t* Vbase = vT + (size_t)kv * HD * S_LEN;

  for (int kt = 0; kt < nch; ++kt) {
    const int t0 = kt * 32;
    v8f s0 = {}, s1 = {};
    const bf16_t* Kp = Kbase + ((size_t)t0 + col) * HD + half * 8;
#pragma unroll
    for (int c4 = 0; c4 < 4; ++c4) {
      v16bf b0 = load_frag(Kp + c4 * 32,           Kp + c4 * 32 + 16);
      v16bf b1 = load_frag(Kp + 16 * HD + c4 * 32, Kp + 16 * HD + c4 * 32 + 16);
      s0 = __builtin_amdgcn_wmma_f32_16x16x32_bf16(false, aq[c4], false, b0, (short)0, s0, false, false);
      s1 = __builtin_amdgcn_wmma_f32_16x16x32_bf16(false, aq[c4], false, b1, (short)0, s1, false, false);
    }
    const int key0 = t0 + col, key1 = t0 + 16 + col;
    float corr[8];
#pragma unroll
    for (int r = 0; r < 8; ++r) {
      const int q = qi * 16 + 8 * half + r;
      float v0 = (key0 <= q) ? s0[r] * sc : -3.0e38f;
      float v1 = (key1 <= q) ? s1[r] * sc : -3.0e38f;
      float mx = fmaxf(v0, v1);
      mx = fmaxf(mx, __shfl_xor(mx, 1, 32));
      mx = fmaxf(mx, __shfl_xor(mx, 2, 32));
      mx = fmaxf(mx, __shfl_xor(mx, 4, 32));
      mx = fmaxf(mx, __shfl_xor(mx, 8, 32));
      const float mnew = fmaxf(mrow[r], mx);
      corr[r] = __expf(mrow[r] - mnew);
      const float p0 = __expf(v0 - mnew);
      const float p1 = __expf(v1 - mnew);
      float ps = p0 + p1;
      ps += __shfl_xor(ps, 1, 32);
      ps += __shfl_xor(ps, 2, 32);
      ps += __shfl_xor(ps, 4, 32);
      ps += __shfl_xor(ps, 8, 32);
      lrow[r] = lrow[r] * corr[r] + ps;
      mrow[r] = mnew;
      // stage P (C-fragment layout) into LDS for A-fragment re-read
      Plds[(8 * half + r) * 32 + col]      = (bf16_t)p0;
      Plds[(8 * half + r) * 32 + 16 + col] = (bf16_t)p1;
    }
#pragma unroll
    for (int dc = 0; dc < 8; ++dc)
#pragma unroll
      for (int r = 0; r < 8; ++r) acc[dc][r] *= corr[r];
    __syncthreads();
    v16bf pf = load_frag(&Plds[col * 32 + half * 8], &Plds[col * 32 + 16 + half * 8]);
#pragma unroll
    for (int dc = 0; dc < 8; ++dc) {
      const bf16_t* Vp = Vbase + ((size_t)dc * 16 + col) * S_LEN + t0 + half * 8;
      v16bf bv = load_frag(Vp, Vp + 16);
      acc[dc] = __builtin_amdgcn_wmma_f32_16x16x32_bf16(false, pf, false, bv, (short)0, acc[dc], false, false);
    }
    __syncthreads();
  }

  // normalize + write ctx as bf16 [s, H*D]
  float inv[8];
#pragma unroll
  for (int r = 0; r < 8; ++r) inv[r] = 1.0f / lrow[r];
#pragma unroll
  for (int r = 0; r < 8; ++r) {
    const int s = qi * 16 + 8 * half + r;
    bf16_t* op = ctxb + (size_t)s * (NH * HD) + h * HD + col;
#pragma unroll
    for (int dc = 0; dc < 8; ++dc) op[dc * 16] = (bf16_t)(acc[dc][r] * inv[r]);
  }
}

extern "C" void kernel_launch(void* const* d_in, const int* in_sizes, int n_in,
                              void* d_out, int out_size, void* d_ws, size_t ws_size,
                              hipStream_t stream) {
  const float* x    = (const float*)d_in[0];
  const float* Wq   = (const float*)d_in[1];
  const float* Wk   = (const float*)d_in[2];
  const float* Wv   = (const float*)d_in[3];
  const float* Wo   = (const float*)d_in[4];
  const float* cosb = (const float*)d_in[5];
  const float* sinb = (const float*)d_in[6];
  float* out = (float*)d_out;

  char* ws = (char*)d_ws;
  size_t off = 0;
  auto alloc = [&](size_t bytes) -> char* {
    char* p = ws + off; off += (bytes + 255) & ~(size_t)255; return p;
  };
  bf16_t* xb    = (bf16_t*)alloc((size_t)S_LEN * HIDW * 2);   // 8 MiB
  bf16_t* wqkvb = (bf16_t*)alloc((size_t)4096 * HIDW * 2);    // 16 MiB (Wq|Wk|Wv rows)
  bf16_t* wob   = (bf16_t*)alloc((size_t)HIDW * HIDW * 2);    // 8 MiB
  float*  qkvf  = (float*)alloc((size_t)S_LEN * 4096 * 4);    // 32 MiB
  bf16_t* qb    = (bf16_t*)alloc((size_t)NH  * S_LEN * HD * 2);
  bf16_t* kbuf  = (bf16_t*)alloc((size_t)NKV * S_LEN * HD * 2);
  bf16_t* vT    = (bf16_t*)alloc((size_t)NKV * S_LEN * HD * 2);
  bf16_t* ctxb  = (bf16_t*)alloc((size_t)S_LEN * HIDW * 2);

  const int T = 256;
  k_f2bf<<<(S_LEN * HIDW + T - 1) / T, T, 0, stream>>>(x, xb, S_LEN * HIDW);
  k_f2bf<<<(2048 * 2048 + T - 1) / T, T, 0, stream>>>(Wq, wqkvb,               2048 * 2048);
  k_f2bf<<<(1024 * 2048 + T - 1) / T, T, 0, stream>>>(Wk, wqkvb + 2048 * 2048, 1024 * 2048);
  k_f2bf<<<(1024 * 2048 + T - 1) / T, T, 0, stream>>>(Wv, wqkvb + 3072 * 2048, 1024 * 2048);
  k_f2bf<<<(2048 * 2048 + T - 1) / T, T, 0, stream>>>(Wo, wob, 2048 * 2048);

  dim3 g1(4096 / 64, 2048 / 32);
  k_gemm_nt<<<g1, 32, 0, stream>>>(xb, wqkvb, qkvf, 2048, 4096, 2048);

  k_rope_pack<<<(S_LEN * 4096 + T - 1) / T, T, 0, stream>>>(qkvf, cosb, sinb, qb, kbuf, vT);

  k_attn<<<NH * (S_LEN / 16), 32, 0, stream>>>(qb, kbuf, vT, ctxb);

  dim3 g2(2048 / 64, 2048 / 32);
  k_gemm_nt<<<g2, 32, 0, stream>>>(ctxb, wob, out, 2048, 2048, 2048);

  (void)in_sizes; (void)n_in; (void)out_size; (void)ws_size;
}
